// MambaFormerBlock_65712999629301
// MI455X (gfx1250) — matrix operand order repaired and checked by reading, not compiled, not run on hardware
//
#include <hip/hip_runtime.h>
#include <hip/hip_bf16.h>
#include <stdint.h>

// ---------------------------------------------------------------------------
// MambaFormer block for MI455X (gfx1250, wave32, WMMA bf16 -> f32 accumulate)
// All WMMA operands live in fragment-swizzled storage (512-element blocks,
// 32 contiguous bytes per lane -> two b128 loads), per ISA 7.12.2 layouts.
// GEMM + attention inner loops are branch-free and (attention) barrier-free.
// ---------------------------------------------------------------------------

typedef __bf16 bf16_t;
typedef __attribute__((ext_vector_type(16))) __bf16 v16bf;
typedef __attribute__((ext_vector_type(8)))  float  v8f;

#define B_SZ    2
#define L_SZ    2048
#define DIM     1024
#define HEADS   16
#define HD      64
#define D_STATE 16
#define D_CONV  4
#define D_INNER 2048
#define DT_RANK 64
#define QKVW    (DIM + 2 * HD)   // 1152
#define BL      (B_SZ * L_SZ)    // 4096
#define KVBLKS  256              // per-batch fragment blocks in Kt / V tables

// -------------------------------- helpers ----------------------------------

__device__ __forceinline__ bf16_t f2bf(float f) {
  union { float f; uint32_t u; } v; v.f = f;
  uint32_t r = v.u + 0x7FFFu + ((v.u >> 16) & 1u);   // round-to-nearest-even
  uint16_t h = (uint16_t)(r >> 16);
  return __builtin_bit_cast(bf16_t, h);
}

__device__ __forceinline__ v8f wmma_bf16(v16bf a, v16bf b, v8f c) {
  return __builtin_amdgcn_wmma_f32_16x16x32_bf16(
      false, a, false, b, (short)0, c, false, false);
}

// Swizzled A storage: [M/16 x K/32] blocks of 512 bf16; within a block the
// element for (m, k) lives at lane*16 + i where (ISA 7.12.2, 16-bit A 16x32):
//   lane = (m&15) + 16*((k>>3)&1) ;  i = 8*((k>>4)&1) + (k&7)
__device__ __forceinline__ size_t a_sw_idx(int m, int k, int K) {
  int lane = (m & 15) + (((k >> 3) & 1) << 4);
  int i    = (((k >> 4) & 1) << 3) + (k & 7);
  size_t blk = (size_t)(m >> 4) * (size_t)(K >> 5) + (size_t)(k >> 5);
  return (blk << 9) + ((size_t)lane << 4) + i;
}

// ------------------------------ cast kernels -------------------------------
// Each thread emits one lane's contiguous 16 bf16 of one fragment block.

__global__ void cast_swizzle_a(const float* __restrict__ src,
                               bf16_t* __restrict__ dst, int M, int K) {
  int t = blockIdx.x * blockDim.x + threadIdx.x;
  int KB = K >> 5;
  int total = (M >> 4) * KB * 32;
  if (t >= total) return;
  int blk = t >> 5, lane = t & 31;
  int mb = blk / KB, kb = blk % KB;
  int m = (mb << 4) + (lane & 15);
  int hbit = (lane >> 4) & 1;           // k bit3
  const float* row = src + (size_t)m * K + ((size_t)kb << 5);
  v16bf o;
#pragma unroll
  for (int i = 0; i < 16; ++i) {
    int k = ((i >> 3) << 4) + (hbit << 3) + (i & 7);
    o[i] = f2bf(row[k]);
  }
  *(v16bf*)(dst + ((size_t)blk << 9) + ((size_t)lane << 4)) = o;
}

// B swizzle with padded column count Npad (multiple of 64); columns >= N are
// zero-filled so the GEMM inner loop never needs a boundary branch.
__global__ void cast_swizzle_b(const float* __restrict__ src,
                               bf16_t* __restrict__ dst, int K, int N, int Npad) {
  int t = blockIdx.x * blockDim.x + threadIdx.x;
  int NB = Npad >> 4;
  int total = (K >> 5) * NB * 32;
  if (t >= total) return;
  int blk = t >> 5, lane = t & 31;
  int kb = blk / NB, nb = blk % NB;
  int n = (nb << 4) + (lane & 15);
  int k0 = (kb << 5) + (((lane >> 4) & 1) << 3);
  v16bf o;
#pragma unroll
  for (int i = 0; i < 16; ++i)
    o[i] = (n < N) ? f2bf(src[(size_t)(k0 + ((i >> 3) << 4) + (i & 7)) * N + n]) : f2bf(0.0f);
  *(v16bf*)(dst + ((size_t)blk << 9) + ((size_t)lane << 4)) = o;
}

// K^T and V fragment tables for MQA attention (B-swizzled, bf16):
//  kt_sw per b: B-matrix [hd=64 x kv=2048] -> 2 x 128 blocks (hd-major)
//  v_sw  per b: B-matrix [kv=2048 x hd=64] -> 64 x 4 blocks  (kv-major)
__global__ void kv_swizzle(const float* __restrict__ qkv,
                           bf16_t* __restrict__ kt_sw,
                           bf16_t* __restrict__ v_sw) {
  int t = blockIdx.x * blockDim.x + threadIdx.x;   // B * 16384 threads
  int b = t >> 14;
  int r = t & 16383;
  int lane = r & 31;
  int blk = (r >> 5) & (KVBLKS - 1);
  bool isV = (r >> 5) >= KVBLKS;
  const float* base = qkv + (size_t)b * L_SZ * QKVW + DIM;
  v16bf o;
  bf16_t* dst;
  if (!isV) {
    // Kt: KB=2 (hd), NB=128 (kv); element (k=hd, n=kv) = qkv[b, kv, DIM+hd]
    int kb = blk >> 7, nb = blk & 127;
    int kv = (nb << 4) + (lane & 15);
    int hd0 = (kb << 5) + (((lane >> 4) & 1) << 3);
    const float* srow = base + (size_t)kv * QKVW + hd0;
#pragma unroll
    for (int i = 0; i < 16; ++i) o[i] = f2bf(srow[((i >> 3) << 4) + (i & 7)]);
    dst = kt_sw + (((size_t)b * KVBLKS + blk) << 9) + ((size_t)lane << 4);
  } else {
    // V: KB=64 (kv), NB=4 (hd); element (k=kv, n=hd) = qkv[b, kv, DIM+HD+hd]
    int kb = blk >> 2, nb = blk & 3;
    int hd = (nb << 4) + (lane & 15);
    int kv0 = (kb << 5) + (((lane >> 4) & 1) << 3);
    const float* scol = base + (size_t)kv0 * QKVW + HD + hd;
#pragma unroll
    for (int i = 0; i < 16; ++i) o[i] = f2bf(scol[(size_t)(((i >> 3) << 4) + (i & 7)) * QKVW]);
    dst = v_sw + (((size_t)b * KVBLKS + blk) << 9) + ((size_t)lane << 4);
  }
  *(v16bf*)dst = o;
}

// ------------------------- generic WMMA GEMM -------------------------------
// LDS-free: fragments load straight from swizzled global (two b128 per frag,
// L2-resident). Block = 128 threads (4 waves); tile 128(M) x 64(N) x 32(K);
// each wave computes 32x64 (2 A-frags x 4 B-frags = 8 WMMA per K-step).

__global__ __launch_bounds__(128) void gemm_bf16_wmma(
    const bf16_t* __restrict__ A,   // swizzled A, K cols
    const bf16_t* __restrict__ Bw,  // swizzled B, Npad cols
    float* __restrict__ C, int ldc,
    bf16_t* __restrict__ Cbf, int mirrorK,       // may be null / 0
    const float* __restrict__ bias,              // may be null
    const float* __restrict__ resid, int ldr,    // may be null
    int N, int Npad, int K) {
  const int n0   = blockIdx.x * 64;
  const int m0   = blockIdx.y * 128;
  const int wave = threadIdx.x >> 5;
  const int lane = threadIdx.x & 31;
  const int half = lane >> 4;
  const int ln   = lane & 15;
  const int KB = K >> 5, NB = Npad >> 4;

  const bf16_t* a0p = A + (((size_t)((m0 >> 4) + wave * 2) * KB) << 9) + ((size_t)lane << 4);
  const bf16_t* a1p = a0p + ((size_t)KB << 9);
  const bf16_t* bp  = Bw + ((size_t)(n0 >> 4) << 9) + ((size_t)lane << 4);

  const v8f zero = {};
  v8f acc[2][4];
#pragma unroll
  for (int g = 0; g < 2; ++g)
#pragma unroll
    for (int nt = 0; nt < 4; ++nt) acc[g][nt] = zero;

  for (int kb = 0; kb < KB; ++kb) {
    const bf16_t* brow = bp + (((size_t)kb * NB) << 9);
    v16bf a0 = *(const v16bf*)(a0p + ((size_t)kb << 9));
    v16bf a1 = *(const v16bf*)(a1p + ((size_t)kb << 9));
    v16bf b0 = *(const v16bf*)(brow);
    v16bf b1 = *(const v16bf*)(brow + 512);
    v16bf b2 = *(const v16bf*)(brow + 1024);
    v16bf b3 = *(const v16bf*)(brow + 1536);
    if (kb + 1 < KB) {
      __builtin_prefetch(a0p + ((size_t)(kb + 1) << 9), 0, 3);
      __builtin_prefetch(a1p + ((size_t)(kb + 1) << 9), 0, 3);
      __builtin_prefetch(bp + (((size_t)(kb + 1) * NB) << 9), 0, 3);
    }
    acc[0][0] = wmma_bf16(a0, b0, acc[0][0]);
    acc[1][0] = wmma_bf16(a1, b0, acc[1][0]);
    acc[0][1] = wmma_bf16(a0, b1, acc[0][1]);
    acc[1][1] = wmma_bf16(a1, b1, acc[1][1]);
    acc[0][2] = wmma_bf16(a0, b2, acc[0][2]);
    acc[1][2] = wmma_bf16(a1, b2, acc[1][2]);
    acc[0][3] = wmma_bf16(a0, b3, acc[0][3]);
    acc[1][3] = wmma_bf16(a1, b3, acc[1][3]);
  }

  // Epilogue: C layout M = r + 8*half, N = ln (per 16-wide frag)
#pragma unroll
  for (int g = 0; g < 2; ++g) {
#pragma unroll
    for (int nt = 0; nt < 4; ++nt) {
#pragma unroll
      for (int r = 0; r < 8; ++r) {
        int gm = m0 + wave * 32 + g * 16 + (half << 3) + r;
        int gn = n0 + nt * 16 + ln;
        if (gn < N) {
          float v = acc[g][nt][r];
          if (bias)  v += bias[gn];
          if (resid) v += resid[(size_t)gm * ldr + gn];
          C[(size_t)gm * ldc + gn] = v;
          if (Cbf && gn < mirrorK) Cbf[a_sw_idx(gm, gn, mirrorK)] = f2bf(v);
        }
      }
    }
  }
}

// --------------------- MQA flash attention (causal) ------------------------
// One block = 4 waves = 64 query rows of one (b, h). K/V fragments stream
// directly from the pre-swizzled global tables (MQA: 16 heads share them, so
// they sit in L2/WGP$). Barrier-free: only per-wave LDS for the P re-swizzle.

__global__ __launch_bounds__(128) void mqa_flash(const float* __restrict__ qkv,
                                                 const bf16_t* __restrict__ kt_sw,
                                                 const bf16_t* __restrict__ v_sw,
                                                 bf16_t* __restrict__ ctx) {
  __shared__ __align__(32) bf16_t sP[4 * 512];

  const int tiles = L_SZ / 64;  // 32
  const int qt = blockIdx.x % tiles;
  const int h  = (blockIdx.x / tiles) % HEADS;
  const int b  = blockIdx.x / (tiles * HEADS);
  const int r0 = qt * 64;

  const int wave = threadIdx.x >> 5;
  const int lane = threadIdx.x & 31;
  const int half = lane >> 4;
  const int ln   = lane & 15;

  const float*  qbase = qkv + (size_t)b * L_SZ * QKVW;
  const bf16_t* ktb = kt_sw + (((size_t)b * KVBLKS) << 9) + ((size_t)lane << 4);
  const bf16_t* vb  = v_sw  + (((size_t)b * KVBLKS) << 9) + ((size_t)lane << 4);

  // Per-wave Q fragments (16 rows x 64 hd), gathered in A-layout from global.
  v16bf aq0, aq1;
  {
    const float* qrow = qbase + (size_t)(r0 + wave * 16 + ln) * QKVW + h * HD;
    int hbit = half << 3;   // k bit3 from lane half
#pragma unroll
    for (int i = 0; i < 16; ++i) {
      int k = ((i >> 3) << 4) + hbit + (i & 7);
      aq0[i] = f2bf(qrow[k]);
      aq1[i] = f2bf(qrow[32 + k]);
    }
  }

  const v8f zero = {};
  v8f o[4];
#pragma unroll
  for (int i = 0; i < 4; ++i) o[i] = zero;
  float mrun[8], lrun[8];
#pragma unroll
  for (int r = 0; r < 8; ++r) { mrun[r] = -3.0e38f; lrun[r] = 0.0f; }

  const int jmax = (r0 + 63) >> 5;
  for (int j = 0; j <= jmax; ++j) {
    const int kv0 = j * 32;
    const int fn  = j * 2;            // kv 16-col chunk index

    // S = Q * K^T : K^T frags at blocks (fk*128 + fn(+1)), fk in {0,1}
    v16bf k00 = *(const v16bf*)(ktb + ((size_t)(fn) << 9));
    v16bf k01 = *(const v16bf*)(ktb + ((size_t)(fn + 1) << 9));
    v16bf k10 = *(const v16bf*)(ktb + ((size_t)(128 + fn) << 9));
    v16bf k11 = *(const v16bf*)(ktb + ((size_t)(128 + fn + 1) << 9));
    if (j < jmax) {
      __builtin_prefetch(ktb + ((size_t)(fn + 2) << 9), 0, 3);
      __builtin_prefetch(ktb + ((size_t)(130 + fn) << 9), 0, 3);
      __builtin_prefetch(vb + ((size_t)((j + 1) * 4) << 9), 0, 3);
    }
    v8f s0 = zero, s1 = zero;
    s0 = wmma_bf16(aq0, k00, s0);
    s0 = wmma_bf16(aq1, k10, s0);
    s1 = wmma_bf16(aq0, k01, s1);
    s1 = wmma_bf16(aq1, k11, s1);

    // Scale + causal mask.
    const float scale = 0.125f;  // 1/sqrt(64)
#pragma unroll
    for (int r = 0; r < 8; ++r) {
      int qrow = r0 + wave * 16 + (half << 3) + r;
      float v0 = s0[r] * scale;
      float v1 = s1[r] * scale;
      if (kv0 + ln > qrow)      v0 = -1.0e30f;
      if (kv0 + 16 + ln > qrow) v1 = -1.0e30f;
      s0[r] = v0; s1[r] = v1;
    }

    // Online softmax (row reductions across the 16-lane half-wave).
    float alpha[8];
#pragma unroll
    for (int r = 0; r < 8; ++r) {
      float mx = fmaxf(s0[r], s1[r]);
#pragma unroll
      for (int off = 1; off < 16; off <<= 1)
        mx = fmaxf(mx, __shfl_xor(mx, off, 32));
      float mnew = fmaxf(mrun[r], mx);
      float a  = __expf(mrun[r] - mnew);
      float p0 = __expf(s0[r] - mnew);
      float p1 = __expf(s1[r] - mnew);
      float sum = p0 + p1;
#pragma unroll
      for (int off = 1; off < 16; off <<= 1)
        sum += __shfl_xor(sum, off, 32);
      lrun[r]  = a * lrun[r] + sum;
      mrun[r]  = mnew;
      alpha[r] = a;
      s0[r] = p0; s1[r] = p1;
    }
#pragma unroll
    for (int nt = 0; nt < 4; ++nt)
#pragma unroll
      for (int r = 0; r < 8; ++r) o[nt][r] *= alpha[r];

    // P (C layout) -> per-wave A-fragment block in LDS (in-order DS, no bar).
    bf16_t* pw = sP + (wave << 9);
#pragma unroll
    for (int r = 0; r < 8; ++r) {
      int m = (half << 3) + r;
      int lt = (m & 15) + (((ln >> 3) & 1) << 4);
      pw[(lt << 4) + (ln & 7)]     = f2bf(s0[r]);   // kcol = ln
      pw[(lt << 4) + 8 + (ln & 7)] = f2bf(s1[r]);   // kcol = 16 + ln
    }
    v16bf pa = *(const v16bf*)(pw + (lane << 4));

    // O += P * V : V frags at blocks (j*4 + nt)
    const bf16_t* vrow = vb + ((size_t)(j * 4) << 9);
#pragma unroll
    for (int nt = 0; nt < 4; ++nt)
      o[nt] = wmma_bf16(pa, *(const v16bf*)(vrow + ((size_t)nt << 9)), o[nt]);
  }

  // Normalize and write A-swizzled bf16 context (M = BL, K = DIM).
#pragma unroll
  for (int r = 0; r < 8; ++r) {
    float inv = 1.0f / lrun[r];
    int gm = b * L_SZ + r0 + wave * 16 + (half << 3) + r;
#pragma unroll
    for (int nt = 0; nt < 4; ++nt)
      ctx[a_sw_idx(gm, h * HD + nt * 16 + ln, DIM)] = f2bf(o[nt][r] * inv);
  }
}

// -------------------- depthwise causal conv (K=4) + SiLU -------------------

__global__ void conv_silu_kernel(const float* __restrict__ xr,
                                 const float* __restrict__ conv_w,
                                 const float* __restrict__ conv_b,
                                 float* __restrict__ u,
                                 bf16_t* __restrict__ u_bf) {
  size_t idx = (size_t)blockIdx.x * blockDim.x + threadIdx.x;
  if (idx >= (size_t)BL * D_INNER) return;
  int d = (int)(idx % D_INNER);
  size_t bl = idx / D_INNER;
  int l = (int)(bl % L_SZ);
  const float* col = xr + (bl - (size_t)l) * (2 * D_INNER) + d;
  float acc = conv_b[d];
#pragma unroll
  for (int t = 0; t < D_CONV; ++t) {
    int ls = l - (D_CONV - 1) + t;
    if (ls >= 0) acc += conv_w[d * D_CONV + t] * col[(size_t)ls * (2 * D_INNER)];
  }
  float s = acc / (1.0f + __expf(-acc));  // silu
  u[idx] = s;
  u_bf[a_sw_idx((int)bl, d, D_INNER)] = f2bf(s);
}

// ------------------------------ softplus -----------------------------------

__global__ void softplus_kernel(float* __restrict__ p, int n) {
  int i = blockIdx.x * blockDim.x + threadIdx.x;
  if (i < n) {
    float x = p[i];
    p[i] = (x > 20.0f) ? x : log1pf(__expf(x));
  }
}

// ------------------------------- SSM scan ----------------------------------

__global__ __launch_bounds__(256) void ssm_scan(
    const float* __restrict__ delta,   // [B, L, 2048]
    const float* __restrict__ xdbl,    // [B, L, 96] (B: 64..79, C: 80..95)
    const float* __restrict__ u,       // [B, L, 2048]
    const float* __restrict__ xr,      // [B, L, 4096] (res = cols 2048..4095)
    const float* __restrict__ A_log,   // [2048, 16]
    const float* __restrict__ Dp,      // [2048]
    bf16_t* __restrict__ y_bf) {       // A-swizzled [BL x 2048]
  const int blocks_per_b = D_INNER / 256;  // 8
  const int b = blockIdx.x / blocks_per_b;
  const int d = (blockIdx.x % blocks_per_b) * 256 + threadIdx.x;

  float Arow[D_STATE];
#pragma unroll
  for (int n = 0; n < D_STATE; ++n) Arow[n] = -__expf(A_log[d * D_STATE + n]);
  const float Dd = Dp[d];

  float h[D_STATE];
#pragma unroll
  for (int n = 0; n < D_STATE; ++n) h[n] = 0.0f;

  __shared__ float sB[D_STATE];
  __shared__ float sC[D_STATE];

  for (int l = 0; l < L_SZ; ++l) {
    const size_t bl = (size_t)b * L_SZ + l;
    if (threadIdx.x < 32) {
      float v = xdbl[bl * 96 + DT_RANK + threadIdx.x];
      if (threadIdx.x < D_STATE) sB[threadIdx.x] = v;
      else                       sC[threadIdx.x - D_STATE] = v;
    }
    __syncthreads();

    float dt = delta[bl * D_INNER + d];
    float uu = u[bl * D_INNER + d];
    float du = dt * uu;
    float y = 0.0f;
#pragma unroll
    for (int n = 0; n < D_STATE; ++n) {
      float dA = __expf(dt * Arow[n]);
      h[n] = dA * h[n] + du * sB[n];
      y += h[n] * sC[n];
    }
    float res = xr[bl * (2 * D_INNER) + D_INNER + d];
    float sres = res / (1.0f + __expf(-res));
    y = (y + uu * Dd) * sres;
    y_bf[a_sw_idx((int)bl, d, D_INNER)] = f2bf(y);
    __syncthreads();
  }
}

// ------------------------------- launcher ----------------------------------

extern "C" void kernel_launch(void* const* d_in, const int* in_sizes, int n_in,
                              void* d_out, int out_size, void* d_ws, size_t ws_size,
                              hipStream_t stream) {
  (void)in_sizes; (void)n_in; (void)out_size; (void)ws_size;

  const float* x        = (const float*)d_in[0];
  const float* wqkv     = (const float*)d_in[1];
  const float* bqkv     = (const float*)d_in[2];
  const float* w_attn_o = (const float*)d_in[3];
  const float* b_attn_o = (const float*)d_in[4];
  const float* w_in     = (const float*)d_in[5];
  const float* conv_w   = (const float*)d_in[6];
  const float* conv_b   = (const float*)d_in[7];
  const float* w_xproj  = (const float*)d_in[8];
  const float* w_dt     = (const float*)d_in[9];
  const float* b_dt     = (const float*)d_in[10];
  const float* A_log    = (const float*)d_in[11];
  const float* Dp       = (const float*)d_in[12];
  const float* w_out    = (const float*)d_in[13];
  float* out = (float*)d_out;

  char* ws = (char*)d_ws;
  size_t off = 0;
  auto alloc = [&](size_t bytes) -> void* {
    off = (off + 255) & ~(size_t)255;
    void* p = ws + off;
    off += bytes;
    return p;
  };

  bf16_t* xbf      = (bf16_t*)alloc((size_t)BL * DIM * 2);
  bf16_t* wqkv_bf  = (bf16_t*)alloc((size_t)DIM * QKVW * 2);
  bf16_t* wao_bf   = (bf16_t*)alloc((size_t)DIM * DIM * 2);
  bf16_t* win_bf   = (bf16_t*)alloc((size_t)DIM * 2 * D_INNER * 2);
  bf16_t* wxp_bf   = (bf16_t*)alloc((size_t)D_INNER * 128 * 2);  // padded N
  bf16_t* wdt_bf   = (bf16_t*)alloc((size_t)DT_RANK * D_INNER * 2);
  bf16_t* wout_bf  = (bf16_t*)alloc((size_t)D_INNER * DIM * 2);
  float*  qkv_f    = (float*)alloc((size_t)BL * QKVW * 4);
  bf16_t* kt_sw    = (bf16_t*)alloc((size_t)B_SZ * KVBLKS * 512 * 2);
  bf16_t* v_sw     = (bf16_t*)alloc((size_t)B_SZ * KVBLKS * 512 * 2);
  bf16_t* ctx_bf   = (bf16_t*)alloc((size_t)BL * DIM * 2);
  float*  x1_f     = (float*)alloc((size_t)BL * DIM * 4);
  bf16_t* x1_bf    = (bf16_t*)alloc((size_t)BL * DIM * 2);
  float*  xr_f     = (float*)alloc((size_t)BL * 2 * D_INNER * 4);
  float*  u_f      = (float*)alloc((size_t)BL * D_INNER * 4);
  bf16_t* u_bf     = (bf16_t*)alloc((size_t)BL * D_INNER * 2);
  float*  xdbl_f   = (float*)alloc((size_t)BL * 96 * 4);
  bf16_t* xdbl_bf  = (bf16_t*)alloc((size_t)BL * DT_RANK * 2);   // mirrorK = 64
  float*  delta_f  = (float*)alloc((size_t)BL * D_INNER * 4);
  bf16_t* y_bf     = (bf16_t*)alloc((size_t)BL * D_INNER * 2);

  auto castA = [&](const float* src, bf16_t* dst, int M, int K) {
    int n = (M >> 4) * (K >> 5) * 32;
    cast_swizzle_a<<<(n + 255) / 256, 256, 0, stream>>>(src, dst, M, K);
  };
  auto castB = [&](const float* src, bf16_t* dst, int K, int N, int Npad) {
    int n = (K >> 5) * (Npad >> 4) * 32;
    cast_swizzle_b<<<(n + 255) / 256, 256, 0, stream>>>(src, dst, K, N, Npad);
  };
  castA(x,        xbf,     BL, DIM);
  castB(wqkv,     wqkv_bf, DIM, QKVW, QKVW);
  castB(w_attn_o, wao_bf,  DIM, DIM, DIM);
  castB(w_in,     win_bf,  DIM, 2 * D_INNER, 2 * D_INNER);
  castB(w_xproj,  wxp_bf,  D_INNER, 96, 128);
  castB(w_dt,     wdt_bf,  DT_RANK, D_INNER, D_INNER);
  castB(w_out,    wout_bf, D_INNER, DIM, DIM);

  // 1) qkv = x @ wqkv + bqkv
  gemm_bf16_wmma<<<dim3(QKVW / 64, BL / 128), 128, 0, stream>>>(
      xbf, wqkv_bf, qkv_f, QKVW, nullptr, 0, bqkv, nullptr, 0,
      QKVW, QKVW, DIM);

  // 2a) swizzle K^T / V fragment tables (bf16)
  kv_swizzle<<<dim3((B_SZ * 16384) / 256), 256, 0, stream>>>(qkv_f, kt_sw, v_sw);

  // 2b) causal MQA flash attention -> ctx (A-swizzled bf16)
  mqa_flash<<<dim3(B_SZ * HEADS * (L_SZ / 64)), 128, 0, stream>>>(
      qkv_f, kt_sw, v_sw, ctx_bf);

  // 3) x1 = ctx @ w_attn_out + b_attn_out + x   (f32 + swizzled bf16 mirror)
  gemm_bf16_wmma<<<dim3(DIM / 64, BL / 128), 128, 0, stream>>>(
      ctx_bf, wao_bf, x1_f, DIM, x1_bf, DIM, b_attn_o, x, DIM,
      DIM, DIM, DIM);

  // 4) xr = x1 @ w_in
  gemm_bf16_wmma<<<dim3((2 * D_INNER) / 64, BL / 128), 128, 0, stream>>>(
      x1_bf, win_bf, xr_f, 2 * D_INNER, nullptr, 0, nullptr, nullptr, 0,
      2 * D_INNER, 2 * D_INNER, DIM);

  // 5) u = silu(depthwise_causal_conv(xr[..., :2048]))
  conv_silu_kernel<<<dim3((BL * D_INNER) / 256), 256, 0, stream>>>(
      xr_f, conv_w, conv_b, u_f, u_bf);

  // 6) x_dbl = u @ w_xproj   (N = 96, B padded to 128; mirror dt-rank cols)
  gemm_bf16_wmma<<<dim3(2, BL / 128), 128, 0, stream>>>(
      u_bf, wxp_bf, xdbl_f, 96, xdbl_bf, DT_RANK, nullptr, nullptr, 0,
      96, 128, D_INNER);

  // 7) delta_in = x_dbl[:, :64] @ w_dt + b_dt
  gemm_bf16_wmma<<<dim3(D_INNER / 64, BL / 128), 128, 0, stream>>>(
      xdbl_bf, wdt_bf, delta_f, D_INNER, nullptr, 0, b_dt, nullptr, 0,
      D_INNER, D_INNER, DT_RANK);

  // 8) delta = softplus(delta_in)  (in place)
  softplus_kernel<<<dim3((BL * D_INNER) / 256), 256, 0, stream>>>(
      delta_f, BL * D_INNER);

  // 9) selective scan -> y (A-swizzled bf16)
  ssm_scan<<<dim3(B_SZ * (D_INNER / 256)), 256, 0, stream>>>(
      delta_f, xdbl_f, u_f, xr_f, A_log, Dp, y_bf);

  // 10) out = y @ w_out + x1
  gemm_bf16_wmma<<<dim3(DIM / 64, BL / 128), 128, 0, stream>>>(
      y_bf, wout_bf, out, DIM, nullptr, 0, nullptr, x1_f, DIM,
      DIM, DIM, D_INNER);
}
